// Seq2SeqSentimentOther_14078902796958
// MI455X (gfx1250) — compile-verified, matrix operand order checked by reading, not run on hardware
//
#include <hip/hip_runtime.h>
#include <hip/hip_bf16.h>
#include <stdint.h>

// ---------------------------------------------------------------------------
// Seq2Seq LSTM (encoder + greedy decoder) for MI455X / gfx1250.
//
// Precision strategy: all GEMMs run through v_wmma_f32_16x16x32_bf16
// (bf16 inputs, f32 accumulate).  Weights converted to bf16 once per launch;
// Wout (32000x1024) shrinks 131MB->65MB and stays resident in the 192MB L2
// across the 64 decode steps (saves ~4GB HBM traffic @ 23.3TB/s).
//
// GEMM micro-kernel: one wave owns a 32x32 output tile (2x2 subtiles, 4 acc).
// K-loop is unrolled by 64 with an explicit ping-pong register double buffer:
// loads write straight into the registers the WMMAs consume (no rotation
// copies), 8 WMMAs + 16 b128 loads per iteration, loads a half-iteration
// ahead of their consumers.  Requires M%32==0, N%32==0, K%64==0 (all shapes
// here satisfy this: M in {32,2048}, N in {512,1024,4096,32000},
// K in {512,1024,1536}).
// ---------------------------------------------------------------------------

typedef __bf16 bf16_t;
typedef __attribute__((ext_vector_type(16))) __bf16 v16bf;
typedef __attribute__((ext_vector_type(2)))  __bf16 bf16x2;
typedef __attribute__((ext_vector_type(8)))  float  v8f;

// ---- fragment loader: 16-bit A/B matrix 16x32 layout (ISA 7.12.2) ----------
// lane L: row = L & 15 ; VGPR r, elem pair (2r,2r+1) holds
//   K = 16*(r>>2) + 2*(r&3) + 8*(L>>4)  (+0 / +1)
// -> per lane this is two contiguous 16B runs; compiler emits 2x b128 loads.
__device__ __forceinline__ v16bf load_frag(const bf16_t* __restrict__ M, int ld,
                                           int row0, int k0, int lane) {
  const int m    = lane & 15;
  const int half = lane >> 4;
  const bf16_t* p = M + (size_t)(row0 + m) * (size_t)ld + k0 + half * 8;
  v16bf f;
#pragma unroll
  for (int r = 0; r < 8; ++r) {
    const int k = ((r >> 2) << 4) + ((r & 3) << 1);  // dword-aligned pair
    bf16x2 w = *reinterpret_cast<const bf16x2*>(p + k);
    f[2 * r]     = w.x;
    f[2 * r + 1] = w.y;
  }
  return f;
}

__device__ __forceinline__ v8f wmma_bf16(v16bf a, v16bf b, v8f c) {
  return __builtin_amdgcn_wmma_f32_16x16x32_bf16(
      /*neg_a=*/false, a, /*neg_b=*/false, b,
      /*c_mod=*/(short)0, c, /*reuse_a=*/false, /*reuse_b=*/false);
}

// ---- generic WMMA GEMM:  Out[M,N] = act( A[M,K] * B[N,K]^T + Cinit + bias )
// A row-major bf16 (lda); B = weight matrix [N,K] row-major bf16 (ldb).
// One 32x32 output tile per wave; 8 waves per 256-thread block.
__global__ void gemm_bf16_wmma(const bf16_t* __restrict__ A, int lda,
                               const bf16_t* __restrict__ Bm, int ldb,
                               const float* __restrict__ Cinit, int ldc,
                               const float* __restrict__ bias,
                               float* __restrict__ Of, bf16_t* __restrict__ Ob,
                               int ldo, int M, int N, int K, int act) {
  const int wave = (int)((blockIdx.x * blockDim.x + threadIdx.x) >> 5);
  const int lane = threadIdx.x & 31;
  const int mt = M >> 5, nt = N >> 5;
  if (wave >= mt * nt) return;               // whole-wave uniform: EXEC stays full
  const int tm = wave % mt, tn = wave / mt;
  const int row0 = tm << 5, col0 = tn << 5;
  const int n = lane & 15, half = lane >> 4;

  v8f acc00, acc01, acc10, acc11;
  if (Cinit) {
#pragma unroll
    for (int v = 0; v < 8; ++v) {
      const int r = v + half * 8;
      acc00[v] = Cinit[(size_t)(row0 + r)      * (size_t)ldc + col0 + n];
      acc01[v] = Cinit[(size_t)(row0 + r)      * (size_t)ldc + col0 + 16 + n];
      acc10[v] = Cinit[(size_t)(row0 + 16 + r) * (size_t)ldc + col0 + n];
      acc11[v] = Cinit[(size_t)(row0 + 16 + r) * (size_t)ldc + col0 + 16 + n];
    }
  } else {
#pragma unroll
    for (int v = 0; v < 8; ++v) {
      acc00[v] = 0.0f; acc01[v] = 0.0f; acc10[v] = 0.0f; acc11[v] = 0.0f;
    }
  }

  // ---- ping-pong double-buffered K loop (K % 64 == 0) ----------------------
  v16bf pa0 = load_frag(A,  lda, row0,      0, lane);
  v16bf pa1 = load_frag(A,  lda, row0 + 16, 0, lane);
  v16bf pb0 = load_frag(Bm, ldb, col0,      0, lane);
  v16bf pb1 = load_frag(Bm, ldb, col0 + 16, 0, lane);
  for (int k0 = 0; k0 < K; k0 += 64) {
    // odd set: k0+32 is always < K because K % 64 == 0
    v16bf qa0 = load_frag(A,  lda, row0,      k0 + 32, lane);
    v16bf qa1 = load_frag(A,  lda, row0 + 16, k0 + 32, lane);
    v16bf qb0 = load_frag(Bm, ldb, col0,      k0 + 32, lane);
    v16bf qb1 = load_frag(Bm, ldb, col0 + 16, k0 + 32, lane);
    acc00 = wmma_bf16(pa0, pb0, acc00);
    acc01 = wmma_bf16(pa0, pb1, acc01);
    acc10 = wmma_bf16(pa1, pb0, acc10);
    acc11 = wmma_bf16(pa1, pb1, acc11);
    if (k0 + 64 < K) {   // even set for next iteration
      pa0 = load_frag(A,  lda, row0,      k0 + 64, lane);
      pa1 = load_frag(A,  lda, row0 + 16, k0 + 64, lane);
      pb0 = load_frag(Bm, ldb, col0,      k0 + 64, lane);
      pb1 = load_frag(Bm, ldb, col0 + 16, k0 + 64, lane);
    }
    acc00 = wmma_bf16(qa0, qb0, acc00);
    acc01 = wmma_bf16(qa0, qb1, acc01);
    acc10 = wmma_bf16(qa1, qb0, acc10);
    acc11 = wmma_bf16(qa1, qb1, acc11);
  }

  if (bias) {
    const float bv0 = bias[col0 + n];
    const float bv1 = bias[col0 + 16 + n];
#pragma unroll
    for (int v = 0; v < 8; ++v) {
      acc00[v] += bv0; acc10[v] += bv0;
      acc01[v] += bv1; acc11[v] += bv1;
    }
  }
  if (act == 1) {
#pragma unroll
    for (int v = 0; v < 8; ++v) {
      acc00[v] = tanhf(acc00[v]); acc01[v] = tanhf(acc01[v]);
      acc10[v] = tanhf(acc10[v]); acc11[v] = tanhf(acc11[v]);
    }
  }
#pragma unroll
  for (int v = 0; v < 8; ++v) {
    const int r = v + half * 8;
    const size_t i00 = (size_t)(row0 + r)      * (size_t)ldo + col0 + n;
    const size_t i01 = (size_t)(row0 + r)      * (size_t)ldo + col0 + 16 + n;
    const size_t i10 = (size_t)(row0 + 16 + r) * (size_t)ldo + col0 + n;
    const size_t i11 = (size_t)(row0 + 16 + r) * (size_t)ldo + col0 + 16 + n;
    if (Of) { Of[i00] = acc00[v]; Of[i01] = acc01[v]; Of[i10] = acc10[v]; Of[i11] = acc11[v]; }
    if (Ob) { Ob[i00] = (bf16_t)acc00[v]; Ob[i01] = (bf16_t)acc01[v];
              Ob[i10] = (bf16_t)acc10[v]; Ob[i11] = (bf16_t)acc11[v]; }
  }
}

// ---- elementwise helpers ---------------------------------------------------
__global__ void cvt_f32_bf16(const float* __restrict__ src, bf16_t* __restrict__ dst,
                             long long rows, int cols, int sld, int dld) {
  long long i = (long long)blockIdx.x * blockDim.x + threadIdx.x;
  if (i >= rows * (long long)cols) return;
  long long r = i / cols;
  int c = (int)(i % cols);
  dst[(size_t)r * dld + c] = (bf16_t)src[(size_t)r * sld + c];
}

__global__ void init_state(float* c, bf16_t* hb, float* cdec,
                           int* last_pred, float* nll_sum, int BH, int B) {
  int i = blockIdx.x * blockDim.x + threadIdx.x;
  if (i < BH) {
    c[i] = 0.0f;
    cdec[i] = 0.0f;
    hb[i] = (bf16_t)0.0f;
  }
  if (i < B) last_pred[i] = 1;  // START_INDEX
  if (i == 0) *nll_sum = 0.0f;
}

__global__ void compute_lengths(const int* __restrict__ sentence, int* lengths,
                                int* last_idx, int B, int T) {
  int b = blockIdx.x * blockDim.x + threadIdx.x;
  if (b >= B) return;
  int cnt = 0;
  for (int t = 0; t < T; ++t) cnt += (sentence[b * T + t] != 0);
  lengths[b] = cnt;
  int li = cnt - 1;
  if (li < 0) li = 0;
  if (li > T - 1) li = T - 1;
  last_idx[b] = li;
}

// emb_tbe[t][b][e] = Wemb_bf[sentence[b][t]][e]
__global__ void embed_all(const int* __restrict__ sentence,
                          const bf16_t* __restrict__ Wemb_bf,
                          bf16_t* __restrict__ emb_tbe, int B, int T, int E) {
  long long i = (long long)blockIdx.x * blockDim.x + threadIdx.x;
  if (i >= (long long)B * T * E) return;
  int e = (int)(i % E);
  long long bt = i / E;
  int b = (int)(bt % B);
  int t = (int)(bt / B);
  emb_tbe[i] = Wemb_bf[(size_t)sentence[b * T + t] * E + e];
}

__global__ void embed_step(const int* __restrict__ last_pred,
                           const bf16_t* __restrict__ Wemb_bf,
                           bf16_t* __restrict__ xh, int B, int E, int ldx) {
  int i = blockIdx.x * blockDim.x + threadIdx.x;
  if (i >= B * E) return;
  int b = i / E, e = i % E;
  xh[(size_t)b * ldx + e] = Wemb_bf[(size_t)last_pred[b] * E + e];
}

// PyTorch gate order i,f,g,o.  c updated in place; h written as bf16 (strided)
// plus optional f32 copy (encoder's hs[t]).
__global__ void lstm_cell_k(const float* __restrict__ G, float* __restrict__ c,
                            bf16_t* __restrict__ hb, int ldhb,
                            float* __restrict__ hf, int B, int H) {
  int i = blockIdx.x * blockDim.x + threadIdx.x;
  if (i >= B * H) return;
  int b = i / H, j = i % H;
  const float* gr = G + (size_t)b * 4 * H;
  float ig = gr[j], fg = gr[H + j], gg = gr[2 * H + j], og = gr[3 * H + j];
  float sI = 1.0f / (1.0f + expf(-ig));
  float sF = 1.0f / (1.0f + expf(-fg));
  float sO = 1.0f / (1.0f + expf(-og));
  float cn = sF * c[i] + sI * tanhf(gg);
  c[i] = cn;
  float hn = sO * tanhf(cn);
  hb[(size_t)b * ldhb + j] = (bf16_t)hn;
  if (hf) hf[i] = hn;
}

__global__ void gather_henc(const float* __restrict__ hs,
                            const int* __restrict__ last_idx,
                            bf16_t* henc_b, int B, int H) {
  int i = blockIdx.x * blockDim.x + threadIdx.x;
  if (i >= B * H) return;
  int b = i / H, j = i % H;
  henc_b[i] = (bf16_t)hs[((size_t)last_idx[b] * B + b) * H + j];
}

// per-(b,t): argmax over V, log-softmax NLL for target, feedback pred.
__global__ void argmax_loss(const float* __restrict__ logits_base,
                            const int* __restrict__ sentence,
                            int* __restrict__ last_pred,
                            float* __restrict__ preds_out,
                            float* __restrict__ nll_sum,
                            int t, int B, int T, int V) {
  int b = blockIdx.x;
  const float* row = logits_base + ((size_t)b * T + t) * V;
  __shared__ float smax[256];
  __shared__ int   sidx[256];
  __shared__ float ssum[256];
  int tid = threadIdx.x;

  float best = -3.0e38f;
  int bidx = V;
  for (int v = tid; v < V; v += 256) {
    float x = row[v];
    if (x > best) { best = x; bidx = v; }
  }
  smax[tid] = best; sidx[tid] = bidx;
  __syncthreads();
  for (int s = 128; s > 0; s >>= 1) {
    if (tid < s) {
      if (smax[tid + s] > smax[tid] ||
          (smax[tid + s] == smax[tid] && sidx[tid + s] < sidx[tid])) {
        smax[tid] = smax[tid + s]; sidx[tid] = sidx[tid + s];
      }
    }
    __syncthreads();
  }
  float mx = smax[0];
  int am = sidx[0];

  float lsum = 0.0f;
  for (int v = tid; v < V; v += 256) lsum += expf(row[v] - mx);
  ssum[tid] = lsum;
  __syncthreads();
  for (int s = 128; s > 0; s >>= 1) {
    if (tid < s) ssum[tid] += ssum[tid + s];
    __syncthreads();
  }

  if (tid == 0) {
    int tgt = sentence[b * T + t];
    float nll = (mx + logf(ssum[0])) - row[tgt];
    if (tgt != 0) atomicAdd(nll_sum, nll);
    last_pred[b] = am;
    preds_out[b * T + t] = (float)am;
  }
}

__global__ void finalize_loss(const int* __restrict__ lengths,
                              const float* __restrict__ nll_sum,
                              float* loss_out, int B) {
  if (blockIdx.x == 0 && threadIdx.x == 0) {
    int s = 0;
    for (int b = 0; b < B; ++b) s += lengths[b];
    float denom = (s > 0) ? (float)s : 1.0f;
    *loss_out = *nll_sum / denom;
  }
}

// ---------------------------------------------------------------------------
extern "C" void kernel_launch(void* const* d_in, const int* in_sizes, int n_in,
                              void* d_out, int out_size, void* d_ws, size_t ws_size,
                              hipStream_t stream) {
  (void)in_sizes; (void)n_in; (void)out_size; (void)ws_size;
  const int V = 32000, E = 512, H = 1024, S = 512, O = 512, B = 32, T = 64;
  const int G4 = 4 * H;            // 4096
  const int KD = E + H;            // 1536 (decoder [x|h] combined K)

  // inputs in setup_inputs() order
  const int*   sentence = (const int*)d_in[0];
  const float* W_emb  = (const float*)d_in[1];
  const float* enc_Wi = (const float*)d_in[2];
  const float* enc_Wh = (const float*)d_in[3];
  const float* enc_b  = (const float*)d_in[4];
  const float* dec_Wi = (const float*)d_in[5];
  const float* dec_Wh = (const float*)d_in[6];
  const float* dec_b  = (const float*)d_in[7];
  const float* Ws     = (const float*)d_in[8];
  const float* bs     = (const float*)d_in[9];
  const float* Wo     = (const float*)d_in[10];
  const float* bo     = (const float*)d_in[11];
  const float* Wc     = (const float*)d_in[12];
  const float* bc     = (const float*)d_in[13];
  const float* Wout   = (const float*)d_in[14];
  const float* bout   = (const float*)d_in[15];

  // workspace carve-up (256B aligned)
  char* wsb = (char*)d_ws;
  size_t off = 0;
  auto alloc = [&](size_t bytes) -> void* {
    off = (off + 255) & ~(size_t)255;
    void* p = wsb + off;
    off += bytes;
    return p;
  };
  bf16_t* Wemb_bf  = (bf16_t*)alloc((size_t)V * E * 2);
  bf16_t* encWi_bf = (bf16_t*)alloc((size_t)G4 * E * 2);
  bf16_t* encWh_bf = (bf16_t*)alloc((size_t)G4 * H * 2);
  bf16_t* decW_bf  = (bf16_t*)alloc((size_t)G4 * KD * 2);   // [Wi | Wh] along K
  bf16_t* Ws_bf    = (bf16_t*)alloc((size_t)S * H * 2);
  bf16_t* Wo_bf    = (bf16_t*)alloc((size_t)O * H * 2);
  bf16_t* Wc_bf    = (bf16_t*)alloc((size_t)H * (S + O) * 2);
  bf16_t* Wout_bf  = (bf16_t*)alloc((size_t)V * H * 2);     // 65MB: L2-resident
  bf16_t* emb_tbe  = (bf16_t*)alloc((size_t)T * B * E * 2);
  float*  Xg       = (float*) alloc((size_t)T * B * G4 * 4);
  float*  hs       = (float*) alloc((size_t)T * B * H * 4);
  float*  Gbuf     = (float*) alloc((size_t)B * G4 * 4);
  float*  cbuf     = (float*) alloc((size_t)B * H * 4);
  bf16_t* hbf      = (bf16_t*)alloc((size_t)B * H * 2);
  float*  cdec     = (float*) alloc((size_t)B * H * 4);
  bf16_t* xh       = (bf16_t*)alloc((size_t)B * KD * 2);    // [x | h_dec]
  bf16_t* catbuf   = (bf16_t*)alloc((size_t)B * (S + O) * 2);
  bf16_t* henc_bf  = (bf16_t*)alloc((size_t)B * H * 2);
  int*    lengths  = (int*)   alloc((size_t)B * 4);
  int*    last_idx = (int*)   alloc((size_t)B * 4);
  int*    last_pred= (int*)   alloc((size_t)B * 4);
  float*  nll_sum  = (float*) alloc(4);

  float* logits_out = (float*)d_out;                       // [B,T,V]
  float* preds_out  = logits_out + (size_t)B * T * V;      // [B,T] as float
  float* loss_out   = preds_out + (size_t)B * T;           // scalar

  auto nblk = [](long long n, int b) { return (unsigned)((n + b - 1) / b); };
  auto gemm_blocks = [](int M, int N) { return (unsigned)(((M / 32) * (N / 32) + 7) / 8); };

  // 1) weight conversion f32 -> bf16
  cvt_f32_bf16<<<nblk((long long)V * E, 256), 256, 0, stream>>>(W_emb,  Wemb_bf,  V,  E, E, E);
  cvt_f32_bf16<<<nblk((long long)G4 * E, 256), 256, 0, stream>>>(enc_Wi, encWi_bf, G4, E, E, E);
  cvt_f32_bf16<<<nblk((long long)G4 * H, 256), 256, 0, stream>>>(enc_Wh, encWh_bf, G4, H, H, H);
  cvt_f32_bf16<<<nblk((long long)G4 * E, 256), 256, 0, stream>>>(dec_Wi, decW_bf,      G4, E, E, KD);
  cvt_f32_bf16<<<nblk((long long)G4 * H, 256), 256, 0, stream>>>(dec_Wh, decW_bf + E,  G4, H, H, KD);
  cvt_f32_bf16<<<nblk((long long)S * H, 256), 256, 0, stream>>>(Ws, Ws_bf, S, H, H, H);
  cvt_f32_bf16<<<nblk((long long)O * H, 256), 256, 0, stream>>>(Wo, Wo_bf, O, H, H, H);
  cvt_f32_bf16<<<nblk((long long)H * (S + O), 256), 256, 0, stream>>>(Wc, Wc_bf, H, S + O, S + O, S + O);
  cvt_f32_bf16<<<nblk((long long)V * H, 256), 256, 0, stream>>>(Wout, Wout_bf, V, H, H, H);

  // 2) state init (graph-replay safe: everything reset every call)
  init_state<<<nblk(B * H, 256), 256, 0, stream>>>(cbuf, hbf, cdec,
                                                   last_pred, nll_sum, B * H, B);
  compute_lengths<<<1, 32, 0, stream>>>(sentence, lengths, last_idx, B, T);

  // 3) embeddings + hoisted encoder input projection:
  //    Xg[T*B, 4H] = emb[T*B, E] @ enc_Wi^T + enc_b     (parallel over all t)
  embed_all<<<nblk((long long)T * B * E, 256), 256, 0, stream>>>(sentence, Wemb_bf,
                                                                 emb_tbe, B, T, E);
  gemm_bf16_wmma<<<gemm_blocks(T * B, G4), 256, 0, stream>>>(
      emb_tbe, E, encWi_bf, E, nullptr, 0, enc_b,
      Xg, nullptr, G4, T * B, G4, E, 0);

  // 4) encoder recurrence: G = Xg[t] + h @ enc_Wh^T ; cell update
  for (int t = 0; t < T; ++t) {
    gemm_bf16_wmma<<<gemm_blocks(B, G4), 256, 0, stream>>>(
        hbf, H, encWh_bf, H, Xg + (size_t)t * B * G4, G4, nullptr,
        Gbuf, nullptr, G4, B, G4, H, 0);
    lstm_cell_k<<<nblk(B * H, 256), 256, 0, stream>>>(
        Gbuf, cbuf, hbf, H, hs + (size_t)t * B * H, B, H);
  }

  // 5) last valid hidden + space transformers (tanh GEMMs)
  gather_henc<<<nblk(B * H, 256), 256, 0, stream>>>(hs, last_idx, henc_bf, B, H);
  gemm_bf16_wmma<<<gemm_blocks(B, S), 256, 0, stream>>>(          // sent_h -> cat[:, :S]
      henc_bf, H, Ws_bf, H, nullptr, 0, bs,
      nullptr, catbuf, S + O, B, S, H, 1);
  gemm_bf16_wmma<<<gemm_blocks(B, O), 256, 0, stream>>>(          // oth_h -> cat[:, S:]
      henc_bf, H, Wo_bf, H, nullptr, 0, bo,
      nullptr, catbuf + S, S + O, B, O, H, 1);
  gemm_bf16_wmma<<<gemm_blocks(B, H), 256, 0, stream>>>(          // dec_h -> xh[:, E:]
      catbuf, S + O, Wc_bf, S + O, nullptr, 0, bc,
      nullptr, xh + E, KD, B, H, S + O, 1);

  // 6) greedy decoder: embed(pred) | h  ->  gates  ->  cell  ->  logits -> argmax
  for (int t = 0; t < T; ++t) {
    embed_step<<<nblk(B * E, 256), 256, 0, stream>>>(last_pred, Wemb_bf, xh, B, E, KD);
    gemm_bf16_wmma<<<gemm_blocks(B, G4), 256, 0, stream>>>(        // fused Wi|Wh, K=1536
        xh, KD, decW_bf, KD, nullptr, 0, dec_b,
        Gbuf, nullptr, G4, B, G4, KD, 0);
    lstm_cell_k<<<nblk(B * H, 256), 256, 0, stream>>>(
        Gbuf, cdec, xh + E, KD, nullptr, B, H);
    gemm_bf16_wmma<<<gemm_blocks(B, V), 256, 0, stream>>>(         // logits (L2-hot Wout)
        xh + E, KD, Wout_bf, H, nullptr, 0, bout,
        logits_out + (size_t)t * V, nullptr, T * V, B, V, H, 0);
    argmax_loss<<<B, 256, 0, stream>>>(logits_out, sentence, last_pred,
                                       preds_out, nll_sum, t, B, T, V);
  }

  // 7) loss = sum(nll * mask) / max(sum(mask), 1)
  finalize_loss<<<1, 32, 0, stream>>>(lengths, nll_sum, loss_out, B);
}
